// GraphAttentionLayer_30975304139053
// MI455X (gfx1250) — matrix operand order repaired and checked by reading, not compiled, not run on hardware
//
#include <hip/hip_runtime.h>
#include <hip/hip_bf16.h>
#include <stdint.h>

typedef __attribute__((ext_vector_type(16))) _Float16 v16h;
typedef __attribute__((ext_vector_type(8)))  _Float16 v8h;
typedef __attribute__((ext_vector_type(8)))  float    v8f;

#define B_    4
#define R_    10
#define N_    64
#define E_    64
#define D_    300
#define H_    512
#define BR_   (B_*R_)       // 40
#define BRN_  (BR_*N_)      // 2560
#define BRNE_ (BRN_*E_)     // 163840
#define SLOPE_ 0.2f

static __device__ __forceinline__ v8f wmma16(v16h a, v16h b, v8f c) {
  // D = A(16x32 f16) * B(32x16 f16) + C(16x16 f32)
  return __builtin_amdgcn_wmma_f32_16x16x32_f16(false, a, false, b, (short)0, c,
                                                false, false);
}

static __device__ __forceinline__ v16h cat8(v8h lo, v8h hi) {
  return __builtin_shufflevector(lo, hi, 0, 1, 2, 3, 4, 5, 6, 7,
                                 8, 9, 10, 11, 12, 13, 14, 15);
}

// ---------------------------------------------------------------------------
// K1a: v1[d] = sum_h Wn[d][h]*a1[h], v2[d] = sum_h Wn[d][h]*a2[h]
// ---------------------------------------------------------------------------
__global__ void gat_vec_kernel(const float* __restrict__ W, const float* __restrict__ a,
                               float* __restrict__ v1, float* __restrict__ v2) {
  int d = threadIdx.x;
  if (d >= D_) return;
  const float* row = W + d * H_;
  float a1acc = 0.f, a2acc = 0.f;
  for (int h = 0; h < H_; ++h) {
    float w = row[h];
    a1acc += w * a[h];
    a2acc += w * a[H_ + h];
  }
  v1[d] = a1acc;
  v2[d] = a2acc;
}

// ---------------------------------------------------------------------------
// K1b: q_proj[br][h] = sum_q ques[br][q] * Wq[q][h]
// ---------------------------------------------------------------------------
__global__ void gat_qproj_kernel(const float* __restrict__ ques, const float* __restrict__ W,
                                 float* __restrict__ qproj) {
  int gid = blockIdx.x * blockDim.x + threadIdx.x;
  if (gid >= BR_ * H_) return;
  int br = gid >> 9;            // /512
  int h  = gid & (H_ - 1);
  const float* q  = ques + br * H_;
  const float* Wq = W + D_ * H_ + h;
  float acc = 0.f;
  for (int k = 0; k < H_; ++k) acc += q[k] * Wq[k * H_];
  qproj[gid] = acc;
}

// ---------------------------------------------------------------------------
// K1c: qa1[br] = q_proj[br]·a1 ; qa2[br] = q_proj[br]·a2
// ---------------------------------------------------------------------------
__global__ void gat_qa_kernel(const float* __restrict__ qproj, const float* __restrict__ a,
                              float* __restrict__ qa1, float* __restrict__ qa2) {
  int t = threadIdx.x;
  if (t >= 2 * BR_) return;
  int br = t % BR_, which = t / BR_;
  const float* q  = qproj + br * H_;
  const float* av = a + which * H_;
  float acc = 0.f;
  for (int h = 0; h < H_; ++h) acc += q[h] * av[h];
  (which ? qa2 : qa1)[br] = acc;
}

// ---------------------------------------------------------------------------
// K1d: s1[brn] = nodes[brn]·v1 + qa1[br]
// ---------------------------------------------------------------------------
__global__ void gat_s1_kernel(const float* __restrict__ nodes, const float* __restrict__ v1,
                              const float* __restrict__ qa1, float* __restrict__ s1) {
  int row = blockIdx.x * blockDim.x + threadIdx.x;
  if (row >= BRN_) return;
  const float* r = nodes + row * D_;
  float acc = 0.f;
  for (int d = 0; d < D_; ++d) acc += r[d] * v1[d];
  s1[row] = acc + qa1[row >> 6];
}

// ---------------------------------------------------------------------------
// K2: w_original[2560x512] = nodes[2560x300] @ Wn[300x512] + q_proj (WMMA f16)
// block = 64x64 output tile, 8 waves, 2 C-tiles per wave sharing the A frag.
// B is staged TRANSPOSED (BsT[n][k]) so every lane's 16 B-halves are
// contiguous -> ds_load_b128 pairs instead of 32x ds_load_u16.
// ---------------------------------------------------------------------------
__global__ __launch_bounds__(256) void gat_worig_kernel(
    const float* __restrict__ A, const float* __restrict__ W,
    const float* __restrict__ qproj, float* __restrict__ wo) {
  __shared__ _Float16 As[64 * 32];    // [m][k] row-major
  __shared__ _Float16 BsT[64 * 32];   // [n][k] (transposed)
  int bm = blockIdx.x >> 3, bn = blockIdx.x & 7;
  int m0 = bm * 64, n0 = bn * 64;
  int tid  = threadIdx.x;
  int lane = tid & 31, wv = tid >> 5;
  int tm = wv >> 1, tn0 = (wv & 1) * 2;
  int mi = lane & 15, hi = lane >> 4;
  int sn  = tid & 63;                 // staging n
  int skb = (tid >> 6) * 8;           // staging k base (8 consecutive k)
  v8f c0 = {}, c1 = {};
  for (int kk = 0; kk < 10; ++kk) {   // K = 300 -> 10 chunks of 32, zero-padded
    int k0 = kk * 32;
    // stage A [m][k]: 8 contiguous halves per thread -> 1 ds_store_b128
#pragma unroll
    for (int j = 0; j < 8; ++j) {
      int l = tid * 8 + j;            // 0..2047
      int ar = l >> 5, ak = l & 31;
      int kg = k0 + ak;
      As[l] = (kg < D_) ? (_Float16)A[(m0 + ar) * D_ + kg] : (_Float16)0.f;
    }
    // stage B transposed [n][k]: lanes read consecutive n (coalesced),
    // each thread packs 8 k-halves and writes one b128
    {
      v8h pk;
#pragma unroll
      for (int j = 0; j < 8; ++j) {
        int kg = k0 + skb + j;
        pk[j] = (kg < D_) ? (_Float16)W[kg * H_ + n0 + sn] : (_Float16)0.f;
      }
      *(v8h*)&BsT[sn * 32 + skb] = pk;
    }
    __syncthreads();
    // fragment gathers: all contiguous, b128 pairs
    const _Float16* Ar = &As[(tm * 16 + mi) * 32];
    v16h a = cat8(*(const v8h*)(Ar + 16 * hi), *(const v8h*)(Ar + 16 * hi + 8));
    const _Float16* Br0 = &BsT[(tn0 * 16 + mi) * 32 + 16 * hi];
    v16h b0 = cat8(*(const v8h*)Br0, *(const v8h*)(Br0 + 8));
    const _Float16* Br1 = &BsT[((tn0 + 1) * 16 + mi) * 32 + 16 * hi];
    v16h b1 = cat8(*(const v8h*)Br1, *(const v8h*)(Br1 + 8));
    c0 = wmma16(a, b0, c0);
    c1 = wmma16(a, b1, c1);
    __syncthreads();
  }
  int col0 = n0 + tn0 * 16 + mi;
  int col1 = col0 + 16;
  float qp0 = qproj[bm * H_ + col0];
  float qp1 = qproj[bm * H_ + col1];
#pragma unroll
  for (int r = 0; r < 8; ++r) {
    int row = m0 + tm * 16 + hi * 8 + r;   // C layout: VGPR r -> M = 8*hi + r
    wo[row * H_ + col0] = c0[r] + qp0;
    wo[row * H_ + col1] = c1[r] + qp1;
  }
}

// ---------------------------------------------------------------------------
// K3 (bandwidth-bound): scores[brne] = leaky( s1[brn] + adj[brne]·v2 + qa2[br] )
// One wave per row; 75 float4 per row split across 32 lanes. v2 is staged into
// LDS with global_load_async_to_lds_b128 (ASYNCcnt path).
// ---------------------------------------------------------------------------
__global__ __launch_bounds__(256) void gat_scores_kernel(
    const float* __restrict__ adj, const float* __restrict__ v2,
    const float* __restrict__ qa2, const float* __restrict__ s1,
    float* __restrict__ sc) {
  __shared__ __align__(16) float v2s[304];
  int tid = threadIdx.x;
  if (tid < 75) {
    unsigned ldsoff = (unsigned)(uintptr_t)(&v2s[tid * 4]);
    const float* gp = v2 + tid * 4;
    asm volatile("global_load_async_to_lds_b128 %0, %1, off"
                 :: "v"(ldsoff), "v"(gp) : "memory");
  }
  asm volatile("s_wait_asynccnt 0" ::: "memory");
  __syncthreads();

  int lane = tid & 31;
  int gw = (blockIdx.x * blockDim.x + tid) >> 5;
  int nw = (gridDim.x * blockDim.x) >> 5;
  const float4* v2f4 = (const float4*)v2s;
  float4 w0 = v2f4[lane];
  float4 w1 = v2f4[lane + 32];
  float4 w2 = (lane < 11) ? v2f4[lane + 64] : make_float4(0.f, 0.f, 0.f, 0.f);
  for (int row = gw; row < BRNE_; row += nw) {
    const float4* rp = (const float4*)(adj + (size_t)row * D_);  // 1200B rows, 16B aligned
    float4 x = rp[lane];
    float acc = x.x * w0.x + x.y * w0.y + x.z * w0.z + x.w * w0.w;
    x = rp[lane + 32];
    acc += x.x * w1.x + x.y * w1.y + x.z * w1.z + x.w * w1.w;
    if (lane < 11) {
      x = rp[lane + 64];
      acc += x.x * w2.x + x.y * w2.y + x.z * w2.z + x.w * w2.w;
    }
#pragma unroll
    for (int off = 16; off > 0; off >>= 1) acc += __shfl_down(acc, off, 32);
    if (lane == 0) {
      float s = s1[row >> 6] + acc + qa2[row >> 12];
      sc[row] = (s > 0.f) ? s : SLOPE_ * s;
    }
  }
}

// ---------------------------------------------------------------------------
// K4: softmax over the R axis (axis=1): columns stride N*E = 4096 floats
// ---------------------------------------------------------------------------
__global__ void gat_softmax_kernel(float* __restrict__ sc) {
  int c = blockIdx.x * blockDim.x + threadIdx.x;
  if (c >= B_ * N_ * E_) return;
  int b = c >> 12, ne = c & 4095;
  float* p = sc + b * (R_ * N_ * E_) + ne;
  float m = -1e30f;
#pragma unroll
  for (int r = 0; r < R_; ++r) m = fmaxf(m, p[r * 4096]);
  float e[R_];
  float sum = 0.f;
#pragma unroll
  for (int r = 0; r < R_; ++r) { e[r] = __expf(p[r * 4096] - m); sum += e[r]; }
  float inv = 1.f / sum;
#pragma unroll
  for (int r = 0; r < R_; ++r) p[r * 4096] = e[r] * inv;
}

// ---------------------------------------------------------------------------
// K5: h_prime = attn(64x64) @ w_original(64x512) per (b,r), fused ELU (WMMA)
// block = one (br, 64-wide h slab); 8 waves, 2 tiles each, K = 64 (2 chunks).
// Bw staged transposed (BwT[n][j]) for contiguous b128 fragment gathers.
// ---------------------------------------------------------------------------
__global__ __launch_bounds__(256) void gat_hprime_kernel(
    const float* __restrict__ attn, const float* __restrict__ wo,
    float* __restrict__ out) {
  __shared__ _Float16 Aat[64 * 64];   // attn [i][j] row-major
  __shared__ _Float16 BwT[64 * 64];   // [n][j] (transposed w_original slab)
  int br = blockIdx.x >> 3, hb = blockIdx.x & 7;
  int h0 = hb * 64;
  int tid = threadIdx.x;
  // stage attn [i][j]: contiguous 16 floats per thread
#pragma unroll
  for (int j = 0; j < 16; ++j) {
    int l = tid * 16 + j;             // 0..4095
    Aat[l] = (_Float16)attn[br * 4096 + l];
  }
  // stage w_original transposed: thread -> fixed n, 16 consecutive j
  {
    int n  = tid & 63;
    int jb = (tid >> 6) * 16;
    v8h p0, p1;
#pragma unroll
    for (int jj = 0; jj < 8; ++jj)
      p0[jj] = (_Float16)wo[(br * 64 + jb + jj) * H_ + h0 + n];
#pragma unroll
    for (int jj = 0; jj < 8; ++jj)
      p1[jj] = (_Float16)wo[(br * 64 + jb + 8 + jj) * H_ + h0 + n];
    *(v8h*)&BwT[n * 64 + jb]     = p0;
    *(v8h*)&BwT[n * 64 + jb + 8] = p1;
  }
  __syncthreads();
  int lane = tid & 31, wv = tid >> 5;
  int tm = wv >> 1, tn0 = (wv & 1) * 2;
  int mi = lane & 15, hi = lane >> 4;
  v8f c0 = {}, c1 = {};
#pragma unroll
  for (int kc = 0; kc < 2; ++kc) {
    int k0 = kc * 32;
    const _Float16* Ar = &Aat[(tm * 16 + mi) * 64 + k0];
    v16h a = cat8(*(const v8h*)(Ar + 16 * hi), *(const v8h*)(Ar + 16 * hi + 8));
    const _Float16* Br0 = &BwT[(tn0 * 16 + mi) * 64 + k0 + 16 * hi];
    v16h b0 = cat8(*(const v8h*)Br0, *(const v8h*)(Br0 + 8));
    const _Float16* Br1 = &BwT[((tn0 + 1) * 16 + mi) * 64 + k0 + 16 * hi];
    v16h b1 = cat8(*(const v8h*)Br1, *(const v8h*)(Br1 + 8));
    c0 = wmma16(a, b0, c0);
    c1 = wmma16(a, b1, c1);
  }
  int col0 = h0 + tn0 * 16 + mi, col1 = col0 + 16;
#pragma unroll
  for (int r = 0; r < 8; ++r) {
    int i = tm * 16 + hi * 8 + r;
    float x0 = c0[r], x1 = c1[r];
    out[(br * 64 + i) * H_ + col0] = (x0 > 0.f) ? x0 : (__expf(x0) - 1.f);
    out[(br * 64 + i) * H_ + col1] = (x1 > 0.f) ? x1 : (__expf(x1) - 1.f);
  }
}

// ---------------------------------------------------------------------------
// Workspace layout (floats): v1@0(320) v2@320(320) qa1@640(40) qa2@680(88)
// qproj@768(20480) s1@21248(2560) w_original@23808(1310720) scores@1334528(163840)
// total = 1,498,368 floats = ~5.72 MB
// ---------------------------------------------------------------------------
extern "C" void kernel_launch(void* const* d_in, const int* in_sizes, int n_in,
                              void* d_out, int out_size, void* d_ws, size_t ws_size,
                              hipStream_t stream) {
  const float* adj   = (const float*)d_in[0];  // (B,R,N,E,D)
  const float* nodes = (const float*)d_in[1];  // (B,R,N,D)
  const float* ques  = (const float*)d_in[2];  // (B,R,H)
  const float* W     = (const float*)d_in[3];  // (D+H, H)
  const float* a     = (const float*)d_in[4];  // (2H, 1)
  float* out = (float*)d_out;                  // (B,R,N,H) f32
  float* ws  = (float*)d_ws;

  float* v1    = ws;            // 300
  float* v2    = ws + 320;      // 300
  float* qa1   = ws + 640;      // 40
  float* qa2   = ws + 680;      // 40
  float* qproj = ws + 768;      // 40*512
  float* s1    = ws + 21248;    // 2560
  float* wo    = ws + 23808;    // 2560*512
  float* sc    = ws + 1334528;  // 163840

  gat_vec_kernel    <<<1,   320, 0, stream>>>(W, a, v1, v2);
  gat_qproj_kernel  <<<80,  256, 0, stream>>>(ques, W, qproj);
  gat_qa_kernel     <<<1,   128, 0, stream>>>(qproj, a, qa1, qa2);
  gat_s1_kernel     <<<10,  256, 0, stream>>>(nodes, v1, qa1, s1);
  gat_worig_kernel  <<<320, 256, 0, stream>>>(nodes, W, qproj, wo);
  gat_scores_kernel <<<2048,256, 0, stream>>>(adj, v2, qa2, s1, sc);
  gat_softmax_kernel<<<64,  256, 0, stream>>>(sc);
  gat_hprime_kernel <<<320, 256, 0, stream>>>(sc, wo, out);
}
